// GRUCell_34943853920831
// MI455X (gfx1250) — matrix operand order repaired and checked by reading, not compiled, not run on hardware
//
#include <hip/hip_runtime.h>
#include <math.h>

// ---------- vector types ----------
typedef __attribute__((ext_vector_type(16))) __bf16        v16bf;
typedef __attribute__((ext_vector_type(2)))  __bf16        bf16x2;
typedef __attribute__((ext_vector_type(8)))  float         v8f;
typedef __attribute__((ext_vector_type(4)))  float         f32x4;
typedef __attribute__((ext_vector_type(4)))  unsigned int  u32x4;

union AFrag {
  v16bf v;
  u32x4 q[2];
};

// ---------- problem sizes (fixed by the reference) ----------
static constexpr int Mdim = 8192;   // batch
static constexpr int Ndim = 1024;   // hidden
static constexpr int Kdim = 1024;   // reduction depth per source

// ---------- tiling ----------
static constexpr int BM   = 128;     // block tile M
static constexpr int BN   = 128;     // block tile N
static constexpr int BK   = 32;      // K chunk == one bf16 WMMA K
static constexpr int ASTR = BK + 8;  // LDS row stride (bf16), 80B: 16B-aligned, staggers banks
static constexpr int BSTR = BK + 8;
static constexpr int KI   = Kdim / BK;  // 32 K-iterations per source

__device__ __forceinline__ unsigned int cvt2_bf16(float a, float b) {
  bf16x2 t;
  t.x = (__bf16)a;
  t.y = (__bf16)b;
  return __builtin_bit_cast(unsigned int, t);
}

__device__ __forceinline__ float frcp(float x) { return __builtin_amdgcn_rcpf(x); }

__device__ __forceinline__ float fsigmoid(float x) {
  return frcp(1.0f + __expf(-x));   // v_exp + v_add + v_rcp
}

__device__ __forceinline__ float ftanh(float x) {
  float e = __expf(2.0f * x);
  return (e - 1.0f) * frcp(e + 1.0f);
}

// Low 32 bits of a generic pointer into LDS == byte offset within LDS aperture.
__device__ __forceinline__ unsigned lds_off(const void* p) {
  return (unsigned)(uintptr_t)p;
}

// CDNA5 async copy, GVS form: mem = SGPR base + per-lane 32-bit byte offset.
// 16B per lane into LDS[lds_byte_off], tracked by ASYNCcnt.
__device__ __forceinline__ void async_copy_b128(unsigned lds_byte_off, unsigned voff,
                                                const unsigned short* sbase) {
  asm volatile("global_load_async_to_lds_b128 %0, %1, %2"
               :: "v"(lds_byte_off), "v"(voff), "s"(sbase)
               : "memory");
}

__device__ __forceinline__ void wait_async0() {
  asm volatile("s_wait_asynccnt 0x0" ::: "memory");
}

// ---------------------------------------------------------------------------
// One-shot fp32 -> bf16 conversion (memory-bound, done once per tensor).
// ---------------------------------------------------------------------------
__global__ __launch_bounds__(256)
void cvt_bf16_kernel(const float* __restrict__ src,
                     unsigned short* __restrict__ dst, int n) {
  int i = (blockIdx.x * 256 + threadIdx.x) * 8;
  if (i >= n) return;
  f32x4 v0 = *(const f32x4*)(src + i);
  f32x4 v1 = *(const f32x4*)(src + i + 4);
  u32x4 p;
  p.x = cvt2_bf16(v0.x, v0.y);
  p.y = cvt2_bf16(v0.z, v0.w);
  p.z = cvt2_bf16(v1.x, v1.y);
  p.w = cvt2_bf16(v1.z, v1.w);
  *(u32x4*)(dst + i) = p;
}

// ---------------------------------------------------------------------------
// bf16 WMMA GEMM (C = A*W^T, W row-major [N,K]) with optional second (A,W)
// pair accumulated into the same C; async double-buffered LDS staging with a
// manually unrolled ping-pong pipeline; fused GRU epilogue:
//   MODE 0: outf = sigmoid(acc)                      (z, fp32)
//   MODE 1: outb = bf16(sigmoid(acc) * hv)           (r .* h, bf16)
//   MODE 2: outf = acc                               (xw, fp32)
//   MODE 3: outf = zv*hv + (1-zv)*tanh(acc + xwv)    (final h')
// ---------------------------------------------------------------------------
template <int MODE>
__global__ __launch_bounds__(256)
void gru_gemm(const unsigned short* __restrict__ A1,
              const unsigned short* __restrict__ W1,
              const unsigned short* __restrict__ A2,
              const unsigned short* __restrict__ W2,
              const float* __restrict__ hv, const float* __restrict__ zv,
              const float* __restrict__ xwv,
              float* __restrict__ outf, unsigned short* __restrict__ outb) {
  __shared__ __align__(16) unsigned short As[2][BM * ASTR];
  __shared__ __align__(16) unsigned short Bs[2][BN * BSTR];

  const int tid  = threadIdx.x;
  const int lane = tid & 31;
  const int wave = tid >> 5;
  const int wm   = wave >> 1;   // 0..3 : wave row  (32 rows each)
  const int wn   = wave & 1;    // 0..1 : wave col  (64 cols each)
  const int half = lane >> 4;   // 0/1
  const int l16  = lane & 15;

  const int m0 = blockIdx.y * BM;
  const int n0 = blockIdx.x * BN;

  // ---- loop-invariant staging addresses ----
  // unit u covers one 16B transfer; 4 units per 32-bf16 row; 512 units per tile
  const int u0 = tid, u1 = tid + 256;
  const int r0 = u0 >> 2, c0 = (u0 & 3) << 3;
  const int r1 = u1 >> 2, c1 = (u1 & 3) << 3;

  const unsigned voffA0 = ((unsigned)(m0 + r0) * Kdim + c0) * 2;  // bytes
  const unsigned voffA1 = ((unsigned)(m0 + r1) * Kdim + c1) * 2;
  const unsigned voffW0 = ((unsigned)(n0 + r0) * Kdim + c0) * 2;
  const unsigned voffW1 = ((unsigned)(n0 + r1) * Kdim + c1) * 2;

  unsigned ldsA[2][2], ldsB[2][2];
#pragma unroll
  for (int bf = 0; bf < 2; ++bf) {
    ldsA[bf][0] = lds_off(&As[bf][r0 * ASTR + c0]);
    ldsA[bf][1] = lds_off(&As[bf][r1 * ASTR + c1]);
    ldsB[bf][0] = lds_off(&Bs[bf][r0 * BSTR + c0]);
    ldsB[bf][1] = lds_off(&Bs[bf][r1 * BSTR + c1]);
  }

  v8f acc[2][4] = {};

  // Issue one K-tile's async copies into LDS buffer `buf` (literal buf index).
  auto stage = [&](int buf, const unsigned short* Ap, const unsigned short* Wp,
                   int kt) {
    const unsigned short* Ab = Ap + kt * BK;   // uniform -> SGPR base
    const unsigned short* Wb = Wp + kt * BK;
    async_copy_b128(ldsA[buf][0], voffA0, Ab);
    async_copy_b128(ldsA[buf][1], voffA1, Ab);
    async_copy_b128(ldsB[buf][0], voffW0, Wb);
    async_copy_b128(ldsB[buf][1], voffW1, Wb);
  };

  // One K-step of fragment loads + 8 WMMAs from LDS buffer `buf`.
  auto compute = [&](int buf) {
    AFrag a[2], b[4];
#pragma unroll
    for (int tm = 0; tm < 2; ++tm) {
      // A 16x32 bf16: lane half selects K group {0..7,16..23} vs {8..15,24..31}
      const unsigned short* base = &As[buf][(wm * 32 + tm * 16 + l16) * ASTR];
      a[tm].q[0] = *(const u32x4*)(base + (half << 3));
      a[tm].q[1] = *(const u32x4*)(base + 16 + (half << 3));
    }
#pragma unroll
    for (int tn = 0; tn < 4; ++tn) {
      // B 32x16 bf16: N = l16, lane half selects K group {0..7,16..23} vs {8..15,24..31} (W row-major [N,K])
      const unsigned short* base =
          &Bs[buf][(wn * 64 + tn * 16 + l16) * BSTR + (half << 3)];
      b[tn].q[0] = *(const u32x4*)(base);
      b[tn].q[1] = *(const u32x4*)(base + 16);
    }
#pragma unroll
    for (int tm = 0; tm < 2; ++tm)
#pragma unroll
      for (int tn = 0; tn < 4; ++tn)
        acc[tm][tn] = __builtin_amdgcn_wmma_f32_16x16x32_bf16(
            false, a[tm].v, false, b[tn].v, (short)0, acc[tm][tn], false, false);
  };

  const int nsrc = (MODE == 0 || MODE == 1) ? 2 : 1;

  for (int s = 0; s < nsrc; ++s) {
    const unsigned short* Ap = s ? A2 : A1;
    const unsigned short* Wp = s ? W2 : W1;

    stage(0, Ap, Wp, 0);
    wait_async0();
    __syncthreads();

    // Ping-pong pipeline, literal buffer indices (no selects, no reg churn).
    for (int kt = 0; kt < KI; kt += 2) {
      if (kt + 1 < KI) stage(1, Ap, Wp, kt + 1);
      compute(0);
      wait_async0();
      __syncthreads();

      if (kt + 2 < KI) stage(0, Ap, Wp, kt + 2);
      compute(1);
      wait_async0();
      __syncthreads();
    }
  }

  // ---- fused epilogue (C layout: VGPR i -> M = i + 8*half, N = l16) ----
#pragma unroll
  for (int tm = 0; tm < 2; ++tm) {
#pragma unroll
    for (int tn = 0; tn < 4; ++tn) {
#pragma unroll
      for (int i = 0; i < 8; ++i) {
        int r = m0 + wm * 32 + tm * 16 + half * 8 + i;
        int c = n0 + wn * 64 + tn * 16 + l16;
        size_t off = (size_t)r * Ndim + c;
        float v = acc[tm][tn][i];
        if (MODE == 0) {
          outf[off] = fsigmoid(v);
        } else if (MODE == 1) {
          __bf16 t16 = (__bf16)(fsigmoid(v) * hv[off]);
          outb[off] = __builtin_bit_cast(unsigned short, t16);
        } else if (MODE == 2) {
          outf[off] = v;
        } else {
          float t = ftanh(v + xwv[off]);
          float z = zv[off];
          outf[off] = z * hv[off] + (1.0f - z) * t;
        }
      }
    }
  }
}

extern "C" void kernel_launch(void* const* d_in, const int* in_sizes, int n_in,
                              void* d_out, int out_size, void* d_ws, size_t ws_size,
                              hipStream_t stream) {
  const float* x   = (const float*)d_in[0];
  const float* h   = (const float*)d_in[1];
  const float* Wf32[6] = {(const float*)d_in[2], (const float*)d_in[3],
                          (const float*)d_in[4], (const float*)d_in[5],
                          (const float*)d_in[6], (const float*)d_in[7]};

  const size_t MN = (size_t)Mdim * Ndim;   // 8M
  const size_t NK = (size_t)Ndim * Kdim;   // 1M

  // ---- workspace layout ----
  unsigned short* xb = (unsigned short*)d_ws;       // x   bf16   (16 MB)
  unsigned short* hb = xb + MN;                     // h   bf16   (16 MB)
  unsigned short* Wb[6];                            // weights bf16 (6 x 2 MB)
  Wb[0] = hb + MN;
  for (int i = 1; i < 6; ++i) Wb[i] = Wb[i - 1] + NK;
  float* zb = (float*)(Wb[5] + NK);                 // z   fp32   (32 MB)
  float* xw = zb + MN;                              // xw  fp32   (32 MB)
  unsigned short* rh = (unsigned short*)(xw + MN);  // r.*h bf16  (16 MB)

  // ---- one-shot conversions ----
  cvt_bf16_kernel<<<dim3((int)(MN / 8 / 256)), dim3(256), 0, stream>>>(x, xb, (int)MN);
  cvt_bf16_kernel<<<dim3((int)(MN / 8 / 256)), dim3(256), 0, stream>>>(h, hb, (int)MN);
  for (int i = 0; i < 6; ++i)
    cvt_bf16_kernel<<<dim3((int)(NK / 8 / 256)), dim3(256), 0, stream>>>(Wf32[i], Wb[i], (int)NK);

  const unsigned short *Wwz = Wb[0], *Wuz = Wb[1], *Wwr = Wb[2], *Wur = Wb[3],
                       *Wu = Wb[4], *Ww = Wb[5];

  dim3 grid(Ndim / BN, Mdim / BM);  // (8, 64)
  dim3 block(256);

  // z = sigmoid(x Wwz^T + h Wuz^T)
  gru_gemm<0><<<grid, block, 0, stream>>>(xb, Wwz, hb, Wuz, nullptr, nullptr, nullptr, zb, nullptr);
  // rh = bf16( sigmoid(x Wwr^T + h Wur^T) .* h )
  gru_gemm<1><<<grid, block, 0, stream>>>(xb, Wwr, hb, Wur, h, nullptr, nullptr, nullptr, rh);
  // xw = x Ww^T
  gru_gemm<2><<<grid, block, 0, stream>>>(xb, Ww, nullptr, nullptr, nullptr, nullptr, nullptr, xw, nullptr);
  // out = z.*h + (1-z).*tanh(rh Wu^T + xw)
  gru_gemm<3><<<grid, block, 0, stream>>>(rh, Wu, nullptr, nullptr, h, zb, xw, (float*)d_out, nullptr);
}
